// PriorGuidedPromptAttention_4750233830143
// MI455X (gfx1250) — matrix operand order repaired and checked by reading, not compiled, not run on hardware
//
#include <hip/hip_runtime.h>
#include <math.h>

// ---------------------------------------------------------------------------
// PriorGuidedPromptAttention for MI455X (gfx1250, wave32, WMMA bf16 16x16x32)
// ---------------------------------------------------------------------------

static constexpr int B_  = 2;
static constexpr int T_  = 2048;
static constexpr int D_  = 1024;
static constexpr int H_  = 16;
static constexpr int HD_ = 64;
static constexpr int P_  = 16;

typedef __attribute__((ext_vector_type(16))) __bf16 v16bf;
typedef __attribute__((ext_vector_type(8)))  __bf16 v8bf;
typedef __attribute__((ext_vector_type(4)))  __bf16 v4bf;
typedef __attribute__((ext_vector_type(8)))  float  v8f;

#define USE_ASYNC_LDS 1

#define WMMA_BF16(A, Bv, C) \
  __builtin_amdgcn_wmma_f32_16x16x32_bf16(false, (A), false, (Bv), (short)0, (C), false, false)

__device__ __forceinline__ float gelu_exact(float x) {
  return 0.5f * x * (1.0f + erff(x * 0.70710678118654752440f));
}

__device__ __forceinline__ v16bf frag_from_halves(v8bf lo, v8bf hi) {
  v16bf a;
#pragma unroll
  for (int j = 0; j < 8; ++j) { a[j] = lo[j]; a[8 + j] = hi[j]; }
  return a;
}

// A fragment (16x32 bf16): row m = L&15, element i<8 -> k = (L>>4)*8+i,
// i>=8 -> k = 16+(L>>4)*8+(i-8). Two 16B loads per lane. `ld` in elements.
__device__ __forceinline__ v16bf load_a_frag(const __bf16* base, int ld) {
  int lane = threadIdx.x & 31;
  const __bf16* row = base + (size_t)(lane & 15) * ld + ((lane >> 4) * 8);
  v8bf lo = *(const v8bf*)(row);
  v8bf hi = *(const v8bf*)(row + 16);
  return frag_from_halves(lo, hi);
}

// B fragment (32x16 bf16) from K-major storage: source row r holds column
// n=r of B with contiguous k. lane n = L&15, element i<8 -> k = (L>>4)*8+i,
// i>=8 -> k = 16+(L>>4)*8+(i-8).
__device__ __forceinline__ v16bf load_b_frag(const __bf16* base, int ld) {
  int lane = threadIdx.x & 31;
  const __bf16* row = base + (size_t)(lane & 15) * ld + ((lane >> 4) * 8);
  v8bf lo = *(const v8bf*)(row);
  v8bf hi = *(const v8bf*)(row + 16);
  return frag_from_halves(lo, hi);
}

// 16-byte global -> LDS copy (CDNA5 async path, tracked by ASYNCcnt).
__device__ __forceinline__ void copy16_g2l(const void* gptr, void* lptr) {
#if USE_ASYNC_LDS
  unsigned lds_off = (unsigned)(unsigned long long)lptr;
  asm volatile("global_load_async_to_lds_b128 %0, %1, off"
               :: "v"(lds_off), "v"(gptr) : "memory");
#else
  *(v8bf*)lptr = *(const v8bf*)gptr;
#endif
}
__device__ __forceinline__ void copy16_wait() {
#if USE_ASYNC_LDS
  asm volatile("s_wait_asynccnt 0x0" ::: "memory");
#endif
}

// ---------------------------------------------------------------------------
// fp32 -> bf16 (4 elements/thread); n must be a multiple of 4 (it is).
// ---------------------------------------------------------------------------
__global__ __launch_bounds__(256) void pga_cvt_kernel(const float* __restrict__ src,
                                                      __bf16* __restrict__ dst, int n) {
  int i4 = (blockIdx.x * 256 + threadIdx.x) * 4;
  if (i4 < n) {
    float4 f = *(const float4*)(src + i4);
    v4bf o = {(__bf16)f.x, (__bf16)f.y, (__bf16)f.z, (__bf16)f.w};
    *(v4bf*)(dst + i4) = o;
  }
}

// fp32 [R,C] -> bf16 transposed [C,R] (weights become K-major for B frags)
__global__ __launch_bounds__(256) void pga_cvt_t_kernel(const float* __restrict__ src,
                                                        __bf16* __restrict__ dst,
                                                        int R, int C) {
  int i = blockIdx.x * 256 + threadIdx.x;
  if (i < R * C) {
    int r = i / C, c = i - r * C;
    dst[(size_t)c * R + r] = (__bf16)src[i];
  }
}

// ---------------------------------------------------------------------------
// NN GEMM: C[M,N] = A[M,K] @ W[K,N] (+bias)(+epilogue).  Requires K % 64 == 0.
// A tile double-buffered in LDS via async global->LDS with STATIC buffer
// parity (K-loop manually unrolled 2x -> no runtime buffer-index math);
// B read directly from pre-transposed K-major weights WT[n][k].
// 64x64 tile, 4 waves. mode: 0=bias, 1=bias+GELU, 2=bias+residual.
// CbT (optional): head-transposed bf16 output vT[b,h,d,t] (V projection).
// ---------------------------------------------------------------------------
__global__ __launch_bounds__(128) void pga_gemm_kernel(
    const __bf16* __restrict__ A, int lda,
    const __bf16* __restrict__ WT, /* [N,K] K-major */
    const float* __restrict__ bias,
    const float* __restrict__ Res,
    float* __restrict__ Cf, __bf16* __restrict__ Cb, __bf16* __restrict__ CbT,
    int ldc, int M, int N, int K, int mode) {
  __shared__ __bf16 As[2][64][48];
  int tid  = threadIdx.x;
  int wave = tid >> 5;
  int lane = tid & 31;
  int hg   = lane >> 4;
  int ln   = lane & 15;
  int m0 = blockIdx.y * 64;
  int n0 = blockIdx.x * 64;

  // per-thread A-staging coordinates (2 chunks of 8 bf16 each), row-clamped
  int ar[2], ac[2];
  const __bf16* agp[2];
#pragma unroll
  for (int cc = 0; cc < 2; ++cc) {
    int chunk = tid + cc * 128;         // 0..255
    ar[cc] = chunk >> 2;                // 0..63
    ac[cc] = (chunk & 3) * 8;           // 0,8,16,24
    int gm = m0 + ar[cc];
    gm = gm < M ? gm : M - 1;           // branch-free clamp (reads stay valid)
    agp[cc] = A + (size_t)gm * lda + ac[cc];
  }

  v8f zero = {0.f, 0.f, 0.f, 0.f, 0.f, 0.f, 0.f, 0.f};
  v8f acc[4];
#pragma unroll
  for (int r = 0; r < 4; ++r) acc[r] = zero;

  const __bf16* wbase = WT + (size_t)(n0 + 16 * wave) * K;
  const __bf16* brow  = wbase + (size_t)ln * K + hg * 16;  // prefetch stream

  auto stage = [&](int kk, int buf) {
#pragma unroll
    for (int cc = 0; cc < 2; ++cc)
      copy16_g2l(agp[cc] + kk, &As[buf][ar[cc]][ac[cc]]);
  };
  auto compute = [&](int kk, int buf) {
    v16bf bfrag = load_b_frag(wbase + kk, K);
#pragma unroll
    for (int r = 0; r < 4; ++r) {
      v16bf afrag = load_a_frag(&As[buf][16 * r][0], 48);
      acc[r] = WMMA_BF16(afrag, bfrag, acc[r]);
    }
  };

  // prologue: K-block 0 into buffer 0
  stage(0, 0);
  copy16_wait();
  __syncthreads();

  for (int k0 = 0; k0 < K; k0 += 64) {
    // k0+32 is always < K (K % 64 == 0): stage into buffer 1 unconditionally
    stage(k0 + 32, 1);
    __builtin_prefetch(brow + k0 + 96, 0, 1);   // B weight stream prefetch
    compute(k0, 0);
    copy16_wait();
    __syncthreads();

    if (k0 + 64 < K) {
      stage(k0 + 64, 0);
      __builtin_prefetch(brow + k0 + 128, 0, 1);
    }
    compute(k0 + 32, 1);
    copy16_wait();
    __syncthreads();
  }

  int gn = n0 + 16 * wave + ln;
  float bval = bias ? bias[gn] : 0.0f;
  auto emit = [&](int gm, float v) {
    v += bval;
    if (mode == 1) v = gelu_exact(v);
    if (mode == 2 && Res) v += Res[(size_t)gm * ldc + gn];
    if (Cf) Cf[(size_t)gm * ldc + gn] = v;
    if (Cb) Cb[(size_t)gm * ldc + gn] = (__bf16)v;
    if (CbT) {
      int bb = gm >> 11, t = gm & (T_ - 1);       // M = B*T rows
      int h = gn >> 6, d = gn & (HD_ - 1);
      CbT[(((size_t)bb * H_ + h) * HD_ + d) * T_ + t] = (__bf16)v;
    }
  };
  if (m0 + 64 <= M) {  // full tile: no per-element guards
#pragma unroll
    for (int r = 0; r < 4; ++r)
#pragma unroll
      for (int p = 0; p < 8; ++p) emit(m0 + 16 * r + p + 8 * hg, acc[r][p]);
  } else {             // edge tile (prompt GEMMs, M=32)
#pragma unroll
    for (int r = 0; r < 4; ++r)
#pragma unroll
      for (int p = 0; p < 8; ++p) {
        int gm = m0 + 16 * r + p + 8 * hg;
        if (gm < M) emit(gm, acc[r][p]);
      }
  }
}

// ---------------------------------------------------------------------------
// Flash attention. ctxf[b,t,h*64+d] (f32) and ctxT[b,h,d,t] (bf16) outputs.
// grid (T/64, H, B), 128 threads; each wave independently owns 16 q-rows.
// K read NT from kb (d-contiguous); V read from head-transposed vT (t-contig).
// No cross-wave LDS -> no barriers; only per-wave P C->A relayout in LDS.
// ---------------------------------------------------------------------------
__global__ __launch_bounds__(128) void pga_flash_kernel(
    const __bf16* __restrict__ qb, const __bf16* __restrict__ kb,
    const __bf16* __restrict__ vT, const float* __restrict__ mask,
    float* __restrict__ ctxf, __bf16* __restrict__ ctxT) {
  __shared__ __bf16 Ps[4][16 * 64];
  int tid  = threadIdx.x;
  int wave = tid >> 5;
  int lane = tid & 31;
  int hg   = lane >> 4;
  int ln   = lane & 15;
  int b = blockIdx.z, h = blockIdx.y;
  int q0 = blockIdx.x * 64 + wave * 16;
  const float scale = 0.125f;  // 1/sqrt(64)

  v16bf qf[2];
  {
    const __bf16* qbase = qb + ((size_t)(b * T_ + q0)) * D_ + h * HD_;
#pragma unroll
    for (int ks = 0; ks < 2; ++ks) qf[ks] = load_a_frag(qbase + ks * 32, D_);
  }
  const __bf16* vbase = vT + ((size_t)(b * H_ + h)) * HD_ * T_;
  const float* mrow = mask + (size_t)b * T_ + ln;

  v8f zero = {0.f, 0.f, 0.f, 0.f, 0.f, 0.f, 0.f, 0.f};
  v8f o[4];
#pragma unroll
  for (int c = 0; c < 4; ++c) o[c] = zero;
  float mrun[8], lrun[8];
#pragma unroll
  for (int p = 0; p < 8; ++p) { mrun[p] = -3.0e38f; lrun[p] = 0.0f; }

  __bf16* Pw = &Ps[wave][0];

  for (int j0 = 0; j0 < T_; j0 += 64) {
    // S = Q K^T for 16x64 block
    v8f s[4];
#pragma unroll
    for (int c = 0; c < 4; ++c) {
      v8f a = zero;
#pragma unroll
      for (int ks = 0; ks < 2; ++ks) {
        const __bf16* kbase =
            kb + ((size_t)(b * T_ + j0 + 16 * c)) * D_ + h * HD_ + ks * 32;
        a = WMMA_BF16(qf[ks], load_b_frag(kbase, D_), a);
      }
      s[c] = a;
    }

    // scale + additive mask + online softmax
    float rmax[8];
#pragma unroll
    for (int p = 0; p < 8; ++p) rmax[p] = mrun[p];
#pragma unroll
    for (int c = 0; c < 4; ++c) {
      float madd = (1.0f - mrow[j0 + 16 * c]) * (-1.0e30f);
#pragma unroll
      for (int p = 0; p < 8; ++p) {
        float sv = s[c][p] * scale + madd;
        s[c][p] = sv;
        rmax[p] = fmaxf(rmax[p], sv);
      }
    }
#pragma unroll
    for (int off = 1; off < 16; off <<= 1)
#pragma unroll
      for (int p = 0; p < 8; ++p)
        rmax[p] = fmaxf(rmax[p], __shfl_xor(rmax[p], off, 32));

    float rsum[8];
#pragma unroll
    for (int p = 0; p < 8; ++p) {
      float al = __expf(mrun[p] - rmax[p]);
      mrun[p] = rmax[p];
      lrun[p] *= al;
      rsum[p] = 0.0f;
#pragma unroll
      for (int c = 0; c < 4; ++c) o[c][p] = o[c][p] * al;
    }
#pragma unroll
    for (int c = 0; c < 4; ++c)
#pragma unroll
      for (int p = 0; p < 8; ++p) {
        float pe = __expf(s[c][p] - rmax[p]);
        s[c][p] = pe;
        rsum[p] += pe;
      }
#pragma unroll
    for (int off = 1; off < 16; off <<= 1)
#pragma unroll
      for (int p = 0; p < 8; ++p) rsum[p] += __shfl_xor(rsum[p], off, 32);
#pragma unroll
    for (int p = 0; p < 8; ++p) lrun[p] += rsum[p];

    // P (C-layout) -> per-wave LDS -> A-layout fragments
#pragma unroll
    for (int c = 0; c < 4; ++c)
#pragma unroll
      for (int p = 0; p < 8; ++p)
        Pw[(p + 8 * hg) * 64 + 16 * c + ln] = (__bf16)s[c][p];

    // O += P @ V ; V fragments contiguous from vT[d][t]
#pragma unroll
    for (int ks = 0; ks < 2; ++ks) {
      v16bf pa = load_a_frag(Pw + ks * 32, 64);
#pragma unroll
      for (int cd = 0; cd < 4; ++cd) {
        v16bf vf = load_b_frag(vbase + (size_t)(16 * cd) * T_ + j0 + ks * 32, T_);
        o[cd] = WMMA_BF16(pa, vf, o[cd]);
      }
    }
  }

  float linv[8];
#pragma unroll
  for (int p = 0; p < 8; ++p) linv[p] = 1.0f / lrun[p];
#pragma unroll
  for (int cd = 0; cd < 4; ++cd)
#pragma unroll
    for (int p = 0; p < 8; ++p) {
      float val = o[cd][p] * linv[p];
      int trow = q0 + p + 8 * hg;
      int dcol = 16 * cd + ln;
      ctxf[((size_t)(b * T_ + trow)) * D_ + h * HD_ + dcol] = val;
      ctxT[(((size_t)b * H_ + h) * HD_ + dcol) * T_ + trow] = (__bf16)val;
    }
}

// ---------------------------------------------------------------------------
// Prompt scores: ps[b,h,p,t] = scale * sum_d prompt[b,p,h*64+d]*q[b,t,h*64+d]
// ---------------------------------------------------------------------------
__global__ __launch_bounds__(128) void pga_pscore_kernel(
    const __bf16* __restrict__ pb, const __bf16* __restrict__ qb,
    float* __restrict__ ps) {
  int tid  = threadIdx.x;
  int wave = tid >> 5;
  int lane = tid & 31;
  int hg   = lane >> 4;
  int ln   = lane & 15;
  int b = blockIdx.z, h = blockIdx.y;
  int t0 = blockIdx.x * 64 + wave * 16;

  v8f acc = {0.f, 0.f, 0.f, 0.f, 0.f, 0.f, 0.f, 0.f};
#pragma unroll
  for (int ks = 0; ks < 2; ++ks) {
    v16bf af = load_a_frag(pb + ((size_t)b * P_) * D_ + h * HD_ + ks * 32, D_);
    v16bf bf = load_b_frag(qb + ((size_t)(b * T_ + t0)) * D_ + h * HD_ + ks * 32, D_);
    acc = WMMA_BF16(af, bf, acc);
  }
#pragma unroll
  for (int p = 0; p < 8; ++p) {
    size_t idx = (((size_t)b * H_ + h) * P_ + p + 8 * hg) * T_ + t0 + ln;
    ps[idx] = acc[p] * 0.125f;
  }
}

// ---------------------------------------------------------------------------
// Double softmax with prior gating (row length T), bf16 out for next GEMM.
// ---------------------------------------------------------------------------
__global__ __launch_bounds__(256) void pga_prior_softmax_kernel(
    const float* __restrict__ ps, const float* __restrict__ prior,
    __bf16* __restrict__ outb) {
  int row = blockIdx.x;
  int b = row / (H_ * P_);
  const float* s  = ps + (size_t)row * T_;
  const float* pr = prior + (size_t)b * T_;
  __bf16* o = outb + (size_t)row * T_;
  __shared__ float red[256];
  int tid = threadIdx.x;

  float v = -3.0e38f;
  for (int t = tid; t < T_; t += 256) v = fmaxf(v, s[t]);
  red[tid] = v; __syncthreads();
  for (int st = 128; st > 0; st >>= 1) {
    if (tid < st) red[tid] = fmaxf(red[tid], red[tid + st]);
    __syncthreads();
  }
  float mx1 = red[0]; __syncthreads();

  v = 0.f;
  for (int t = tid; t < T_; t += 256) v += __expf(s[t] - mx1);
  red[tid] = v; __syncthreads();
  for (int st = 128; st > 0; st >>= 1) {
    if (tid < st) red[tid] += red[tid + st];
    __syncthreads();
  }
  float inv1 = 1.0f / red[0]; __syncthreads();

  v = -3.0e38f;
  for (int t = tid; t < T_; t += 256) {
    float y = __expf(s[t] - mx1) * inv1 * pr[t];
    v = fmaxf(v, y);
  }
  red[tid] = v; __syncthreads();
  for (int st = 128; st > 0; st >>= 1) {
    if (tid < st) red[tid] = fmaxf(red[tid], red[tid + st]);
    __syncthreads();
  }
  float mx2 = red[0]; __syncthreads();

  v = 0.f;
  for (int t = tid; t < T_; t += 256) {
    float y = __expf(s[t] - mx1) * inv1 * pr[t];
    v += __expf(y - mx2);
  }
  red[tid] = v; __syncthreads();
  for (int st = 128; st > 0; st >>= 1) {
    if (tid < st) red[tid] += red[tid + st];
    __syncthreads();
  }
  float inv2 = 1.0f / red[0]; __syncthreads();

  for (int t = tid; t < T_; t += 256) {
    float y = __expf(s[t] - mx1) * inv1 * pr[t];
    o[t] = (__bf16)(__expf(y - mx2) * inv2);
  }
}

// ---------------------------------------------------------------------------
// p_v[b,p,h*64+d] = sum_t pa[b,h,p,t] * ctx[b,h,t,d]; ctx from ctxT (t-contig)
// grid (H, B), 128 threads; LDS-free, barrier-free.
// ---------------------------------------------------------------------------
__global__ __launch_bounds__(128) void pga_pv_kernel(
    const __bf16* __restrict__ pa, const __bf16* __restrict__ ctxT,
    float* __restrict__ pv) {
  int tid  = threadIdx.x;
  int wave = tid >> 5;
  int lane = tid & 31;
  int hg   = lane >> 4;
  int ln   = lane & 15;
  int h = blockIdx.x, b = blockIdx.y;

  const __bf16* abase = pa + (((size_t)b * H_ + h) * P_) * T_;
  const __bf16* bbase = ctxT + (((size_t)b * H_ + h) * HD_ + 16 * wave) * T_;
  v8f acc = {0.f, 0.f, 0.f, 0.f, 0.f, 0.f, 0.f, 0.f};
  for (int k0 = 0; k0 < T_; k0 += 32) {
    v16bf af = load_a_frag(abase + k0, T_);
    v16bf bf = load_b_frag(bbase + k0, T_);
    acc = WMMA_BF16(af, bf, acc);
  }
#pragma unroll
  for (int p = 0; p < 8; ++p) {
    size_t idx = ((size_t)b * P_ + p + 8 * hg) * D_ + h * HD_ + 16 * wave + ln;
    pv[idx] = acc[p];
  }
}

// ---------------------------------------------------------------------------
// LayerNorm over D=1024: h = Xa (+Xb); optional f32 h out; bf16 LN out.
// One block per row, float4 per thread.
// ---------------------------------------------------------------------------
__global__ __launch_bounds__(256) void pga_ln_kernel(
    const float* __restrict__ Xa, const float* __restrict__ Xb,
    const float* __restrict__ g, const float* __restrict__ bt,
    float* __restrict__ sum_out, __bf16* __restrict__ outb) {
  int row = blockIdx.x;
  int tid = threadIdx.x;
  const float* xa = Xa + (size_t)row * D_;
  __shared__ float red[256];

  float4 t4 = ((const float4*)xa)[tid];
  if (Xb) {
    float4 u4 = ((const float4*)(Xb + (size_t)row * D_))[tid];
    t4.x += u4.x; t4.y += u4.y; t4.z += u4.z; t4.w += u4.w;
  }
  float ls = t4.x + t4.y + t4.z + t4.w;
  red[tid] = ls; __syncthreads();
  for (int st = 128; st > 0; st >>= 1) {
    if (tid < st) red[tid] += red[tid + st];
    __syncthreads();
  }
  float mean = red[0] * (1.0f / D_); __syncthreads();

  float dx = t4.x - mean, dy = t4.y - mean, dz = t4.z - mean, dw = t4.w - mean;
  red[tid] = dx * dx + dy * dy + dz * dz + dw * dw; __syncthreads();
  for (int st = 128; st > 0; st >>= 1) {
    if (tid < st) red[tid] += red[tid + st];
    __syncthreads();
  }
  float rstd = rsqrtf(red[0] * (1.0f / D_) + 1e-6f);

  int c = tid * 4;
  if (sum_out) ((float4*)(sum_out + (size_t)row * D_))[tid] = t4;
  float4 g4 = ((const float4*)g)[tid];
  float4 b4 = ((const float4*)bt)[tid];
  v4bf y = {(__bf16)(dx * rstd * g4.x + b4.x), (__bf16)(dy * rstd * g4.y + b4.y),
            (__bf16)(dz * rstd * g4.z + b4.z), (__bf16)(dw * rstd * g4.w + b4.w)};
  *(v4bf*)(outb + (size_t)row * D_ + c) = y;
}

// ---------------------------------------------------------------------------
// Host orchestration
// ---------------------------------------------------------------------------
extern "C" void kernel_launch(void* const* d_in, const int* in_sizes, int n_in,
                              void* d_out, int out_size, void* d_ws, size_t ws_size,
                              hipStream_t stream) {
  (void)in_sizes; (void)n_in; (void)out_size; (void)ws_size;

  const float* x      = (const float*)d_in[0];
  const float* p1     = (const float*)d_in[1];
  const float* p2     = (const float*)d_in[2];
  const float* prior1 = (const float*)d_in[3];
  const float* prior2 = (const float*)d_in[4];
  const float* mask   = (const float*)d_in[5];
  const float* Wq  = (const float*)d_in[6];
  const float* bq  = (const float*)d_in[7];
  const float* Wk  = (const float*)d_in[8];
  const float* bk  = (const float*)d_in[9];
  const float* Wv  = (const float*)d_in[10];
  const float* bv  = (const float*)d_in[11];
  const float* g   = (const float*)d_in[12];
  const float* be  = (const float*)d_in[13];
  const float* W1  = (const float*)d_in[14];
  const float* b1  = (const float*)d_in[15];
  const float* W2  = (const float*)d_in[16];
  const float* b2  = (const float*)d_in[17];
  const float* Wp1 = (const float*)d_in[18];
  const float* bp1 = (const float*)d_in[19];
  const float* Wp2 = (const float*)d_in[20];
  const float* bp2 = (const float*)d_in[21];

  const size_t NTD = (size_t)B_ * T_ * D_;      // 4194304
  const size_t DD  = (size_t)D_ * D_;           // 1048576
  const size_t BPD = (size_t)B_ * P_ * D_;      // 32768
  const size_t PS  = (size_t)B_ * H_ * P_ * T_; // 1048576

  char* wsp = (char*)d_ws;
  auto take = [&](size_t bytes) -> char* {
    char* p = wsp;
    wsp += (bytes + 255) & ~(size_t)255;
    return p;
  };

  __bf16* xb    = (__bf16*)take(NTD * 2);
  __bf16* wqt   = (__bf16*)take(DD * 2);   // all weights stored [N,K] K-major
  __bf16* wkt   = (__bf16*)take(DD * 2);
  __bf16* wvt   = (__bf16*)take(DD * 2);
  __bf16* w1t   = (__bf16*)take(DD * 2);
  __bf16* w2t   = (__bf16*)take(DD * 2);
  __bf16* wp1t  = (__bf16*)take(DD * 2);
  __bf16* wp2t  = (__bf16*)take(DD * 2);
  __bf16* p1bb  = (__bf16*)take(BPD * 2);
  __bf16* p2bb  = (__bf16*)take(BPD * 2);
  __bf16* qb2   = (__bf16*)take(NTD * 2);
  __bf16* kb2   = (__bf16*)take(NTD * 2);
  __bf16* vtb   = (__bf16*)take(NTD * 2);  // v, head-transposed [b,h,d,t]
  float*  ctxf  = (float*)take(NTD * 4);
  __bf16* ctxtb = (__bf16*)take(NTD * 2);  // ctx, head-transposed [b,h,d,t]
  float*  resid = (float*)take(NTD * 4);
  __bf16* ln1b  = (__bf16*)take(NTD * 2);
  __bf16* ff1b  = (__bf16*)take(NTD * 2);
  float*  p1sf  = (float*)take(PS * 4);
  float*  p2sf  = (float*)take(PS * 4);
  __bf16* p1ab  = (__bf16*)take(PS * 2);
  __bf16* p2ab  = (__bf16*)take(PS * 2);
  float*  p1v   = (float*)take(BPD * 4);
  float*  p2v   = (float*)take(BPD * 4);
  __bf16* p1lnb = (__bf16*)take(BPD * 2);
  __bf16* p2lnb = (__bf16*)take(BPD * 2);

  auto cvt = [&](const float* s, __bf16* d, size_t n) {
    pga_cvt_kernel<<<(unsigned)((n / 4 + 255) / 256), 256, 0, stream>>>(s, d, (int)n);
  };
  auto cvt_t = [&](const float* s, __bf16* d) {
    pga_cvt_t_kernel<<<(unsigned)((DD + 255) / 256), 256, 0, stream>>>(s, d, D_, D_);
  };

  // 1. precision staging (weights transposed to K-major)
  cvt(x, xb, NTD);
  cvt_t(Wq, wqt);  cvt_t(Wk, wkt);  cvt_t(Wv, wvt);
  cvt_t(W1, w1t);  cvt_t(W2, w2t);
  cvt_t(Wp1, wp1t); cvt_t(Wp2, wp2t);
  cvt(p1, p1bb, BPD); cvt(p2, p2bb, BPD);

  const int M = B_ * T_;  // 4096
  dim3 ggrid(D_ / 64, M / 64);

  // 2. QKV projections (V written head-transposed for flash P@V)
  pga_gemm_kernel<<<ggrid, 128, 0, stream>>>(xb, D_, wqt, bq, nullptr,
                                             nullptr, qb2, nullptr, D_, M, D_, D_, 0);
  pga_gemm_kernel<<<ggrid, 128, 0, stream>>>(xb, D_, wkt, bk, nullptr,
                                             nullptr, kb2, nullptr, D_, M, D_, D_, 0);
  pga_gemm_kernel<<<ggrid, 128, 0, stream>>>(xb, D_, wvt, bv, nullptr,
                                             nullptr, nullptr, vtb, D_, M, D_, D_, 0);

  // 3. flash attention -> ctx (f32 merged + bf16 head-transposed)
  pga_flash_kernel<<<dim3(T_ / 64, H_, B_), 128, 0, stream>>>(qb2, kb2, vtb,
                                                              mask, ctxf, ctxtb);

  // 4. prompt scores + prior-gated double softmax
  pga_pscore_kernel<<<dim3(T_ / 64, H_, B_), 128, 0, stream>>>(p1bb, qb2, p1sf);
  pga_pscore_kernel<<<dim3(T_ / 64, H_, B_), 128, 0, stream>>>(p2bb, qb2, p2sf);
  pga_prior_softmax_kernel<<<B_ * H_ * P_, 256, 0, stream>>>(p1sf, prior1, p1ab);
  pga_prior_softmax_kernel<<<B_ * H_ * P_, 256, 0, stream>>>(p2sf, prior2, p2ab);

  // 5. prompt context: p_v = p_a @ ctx
  pga_pv_kernel<<<dim3(H_, B_), 128, 0, stream>>>(p1ab, ctxtb, p1v);
  pga_pv_kernel<<<dim3(H_, B_), 128, 0, stream>>>(p2ab, ctxtb, p2v);

  // 6. main tail: resid = x+ctx ; LN ; GELU(@W1+b1) ; @W2+b2+resid
  pga_ln_kernel<<<M, 256, 0, stream>>>(x, ctxf, g, be, resid, ln1b);
  pga_gemm_kernel<<<ggrid, 128, 0, stream>>>(ln1b, D_, w1t, b1, nullptr,
                                             nullptr, ff1b, nullptr, D_, M, D_, D_, 1);
  pga_gemm_kernel<<<ggrid, 128, 0, stream>>>(ff1b, D_, w2t, b2, resid,
                                             (float*)d_out, nullptr, nullptr, D_, M, D_, D_, 2);

  // 7. prompt tails: LN(p_v + p) ; GELU(@Wp + bp)
  float* out_p1 = (float*)d_out + NTD;
  float* out_p2 = out_p1 + BPD;
  dim3 pgrid(D_ / 64, 1);
  pga_ln_kernel<<<B_ * P_, 256, 0, stream>>>(p1v, p1, g, be, nullptr, p1lnb);
  pga_gemm_kernel<<<pgrid, 128, 0, stream>>>(p1lnb, D_, wp1t, bp1, nullptr,
                                             out_p1, nullptr, nullptr, D_, B_ * P_, D_, D_, 1);
  pga_ln_kernel<<<B_ * P_, 256, 0, stream>>>(p2v, p2, g, be, nullptr, p2lnb);
  pga_gemm_kernel<<<pgrid, 128, 0, stream>>>(p2lnb, D_, wp2t, bp2, nullptr,
                                             out_p2, nullptr, nullptr, D_, B_ * P_, D_, D_, 1);
}
